// LSTM_77584289235419
// MI455X (gfx1250) — matrix operand order repaired and checked by reading, not compiled, not run on hardware
//
#include <hip/hip_runtime.h>
#include <hip/hip_bf16.h>

// ---------------------------------------------------------------------------
// Fused autoregressive LSTM forecaster for MI455X (gfx1250, wave32, WMMA).
//   B=512, HIST=24, C=184, F=17, PRED=24, HID=32 -> N = 94208 rows
// fc_in folded into the gate GEMM (no nonlinearity between them):
//   gates = x @ (W_ih*W_in)^T + h @ W_hh^T + (W_ih*b_in + b_ih + b_hh)
// One wave32 owns a 16-row tile; per step: one K=96 WMMA pass
// (24 x v_wmma_f32_16x16x32_bf16), v_tanh-based pointwise, shfl fc_out.
// ---------------------------------------------------------------------------

#define BSZ   512
#define HIST  24
#define CCH   184
#define FDIM  17
#define PRED  24
#define HID   32
#define NROWS (BSZ * CCH)        // 94208 = 16 * 5888 = 64 * 1472 (exact)
#define XK    96                 // gate K: 41 x | 7 pad | 32 h(interleaved) | 16 pad
#define FSTRIDE (CCH * FDIM)     // feature stride per timestep (3128 floats)

typedef __bf16 bf16_t;
typedef __attribute__((ext_vector_type(16))) __bf16 v16bf;
typedef __attribute__((ext_vector_type(8)))  __bf16 v8bf;   // 16 bytes
typedef __attribute__((ext_vector_type(8)))  float  v8f;

union ABPack { v16bf full; v8bf half[2]; };
struct __attribute__((packed, aligned(4))) bf16x2 { bf16_t lo, hi; };

// CDNA5 has a 1-instruction V_TANH_F32 trans op; fall back to exp-based form.
#if __has_builtin(__builtin_amdgcn_tanhf)
  #define TANHF(x) __builtin_amdgcn_tanhf(x)
#elif __has_builtin(__builtin_amdgcn_tanh_f32)
  #define TANHF(x) __builtin_amdgcn_tanh_f32(x)
#else
  #define TANHF(x) (1.0f - 2.0f * __builtin_amdgcn_rcpf(__expf(2.0f * (x)) + 1.0f))
#endif
__device__ __forceinline__ float fast_sigmoid(float x) {
  return __builtin_fmaf(0.5f, TANHF(0.5f * x), 0.5f);
}

#if __has_builtin(__builtin_amdgcn_sched_barrier)
  #define SCHED_FENCE() __builtin_amdgcn_sched_barrier(0)
#else
  #define SCHED_FENCE() do {} while (0)
#endif

// A operand (16x32 K-chunk at kbase) from bf16 LDS tile [16][XK].
// ISA 7.12.2 16-bit A layout: lane<16 -> row=lane, K {kb..kb+7, kb+16..kb+23};
// lane>=16 -> row=lane-16, K {kb+8..kb+15, kb+24..kb+31}.
__device__ __forceinline__ v16bf lds_load_A(const bf16_t* tile, int kbase, int lane) {
  const int half = lane >> 4;
  const int row  = lane & 15;
  const bf16_t* p = tile + row * XK + kbase + half * 8;
  ABPack a;
  a.half[0] = *(const v8bf*)p;
  a.half[1] = *(const v8bf*)(p + 16);
  return a.full;
}

// B operand (32x16 K-chunk at kb, out cols n0..n0+15) from bf16 LDS [out][XK].
// lane<16 -> col=lane, K {kb..kb+7, kb+16..kb+23}; lane>=16 -> col=lane-16, K {kb+8..kb+15, kb+24..kb+31}.
__device__ __forceinline__ v16bf lds_load_B(const bf16_t* w, int n0, int kb, int lane) {
  const int half = lane >> 4;
  const int col  = lane & 15;
  const bf16_t* p = w + (n0 + col) * XK + kb + half * 8;
  ABPack b;
  b.half[0] = *(const v8bf*)p;
  b.half[1] = *(const v8bf*)(p + 16);
  return b.full;
}

#define WMMA_BF16(A, B, C) \
  __builtin_amdgcn_wmma_f32_16x16x32_bf16(false, (A), false, (B), (short)0, (C), false, false)

__global__ __launch_bounds__(128)
void lstm_fused_kernel(const float* __restrict__ pm25,
                       const float* __restrict__ feat,
                       const float* __restrict__ Win,  const float* __restrict__ bin,
                       const float* __restrict__ Wih,  const float* __restrict__ Whh,
                       const float* __restrict__ bih,  const float* __restrict__ bhh,
                       const float* __restrict__ Wout, const float* __restrict__ bout,
                       float* __restrict__ out)
{
  // LDS: fused gate weights [128][96] bf16 (24 KB) + per-wave staging (12 KB).
  __shared__ __attribute__((aligned(16))) bf16_t s_wg[128][XK];
  __shared__ __attribute__((aligned(16))) bf16_t s_x [4][16][XK];
  __shared__ float s_bg[128];

  const int tid  = threadIdx.x;
  const int lane = tid & 31;
  const int wave = tid >> 5;
  const int row0 = (blockIdx.x * 4 + wave) * 16;

  // ---- one-time fused weight build: thread tid owns gate row g = tid ----
  {
    const int g = tid;
    float wih[HID];
    #pragma unroll
    for (int m = 0; m < HID; ++m) wih[m] = Wih[g * HID + m];
    // x-path: Wcomb[g][j] = sum_m Wih[g][m] * Win[m][j]   (j < 41)
    for (int j = 0; j < HIST + FDIM; ++j) {
      float s = 0.0f;
      #pragma unroll
      for (int m = 0; m < HID; ++m) s += wih[m] * Win[m * (HIST + FDIM) + j];
      s_wg[g][j] = (bf16_t)s;
    }
    for (int j = HIST + FDIM; j < 48; ++j) s_wg[g][j] = (bf16_t)0.0f;
    // h-path, K-interleaved pairs: logical k -> K = 48 + 2*(k&15) + (k>>4)
    #pragma unroll
    for (int k = 0; k < HID; ++k)
      s_wg[g][48 + 2 * (k & 15) + (k >> 4)] = (bf16_t)Whh[g * HID + k];
    for (int j = 80; j < XK; ++j) s_wg[g][j] = (bf16_t)0.0f;
    // fused bias: b_ih + b_hh + Wih @ b_in
    float b = bih[g] + bhh[g];
    #pragma unroll
    for (int m = 0; m < HID; ++m) b += wih[m] * bin[m];
    s_bg[g] = b;
  }

  // ---- init per-wave staging: xn history + zero pads / h0 ----
  bf16_t* xs = &s_x[wave][0][0];
  for (int e = lane; e < 16 * HIST; e += 32) {
    int r = e / HIST, j = e % HIST;
    xs[r * XK + j] = (bf16_t)pm25[(size_t)(row0 + r) * HIST + j];
  }
  for (int e = lane; e < 16 * (XK - HIST); e += 32) {   // cols 24..95 = 0
    int r = e / (XK - HIST), j = e % (XK - HIST);
    xs[r * XK + HIST + j] = (bf16_t)0.0f;
  }
  __syncthreads();

  // ---- per-lane constants (C/D layout: lane holds col = lane&15) ----
  const int ncol  = lane & 15;
  const int rhalf = (lane >> 4) * 8;
  float bias_g[8];
  #pragma unroll
  for (int nt = 0; nt < 8; ++nt) bias_g[nt] = s_bg[nt * 16 + ncol];
  const float wo0 = Wout[ncol], wo1 = Wout[16 + ncol];
  const float bo  = bout[0];

  // ---- hoisted address math (all loop-invariant); int offsets -> GVS loads ----
  int foff[9], xoff[9];
  #pragma unroll
  for (int i = 0; i < 9; ++i) {
    int e = lane + i * 32;
    if (e < 16 * FDIM) {
      int r = e / FDIM, j = e % FDIM;
      int n = row0 + r;
      int b = n / CCH, c = n % CCH;
      foff[i] = ((b * (HIST + PRED) + HIST) * CCH + c) * FDIM + j;
      xoff[i] = r * XK + HIST + j;
    } else { foff[i] = -1; xoff[i] = 0; }
  }
  int sh_rd[12];
  #pragma unroll
  for (int i = 0; i < 12; ++i) {
    int e = lane + i * 32;
    sh_rd[i] = (e < 16 * 23) ? ((e / 23) * XK + (e % 23) + 1) : -1;
  }
  int obase[8];
  #pragma unroll
  for (int v = 0; v < 8; ++v) {
    int n = row0 + rhalf + v;
    int b = n / CCH, c = n % CCH;
    obase[v] = b * PRED * CCH + c;
  }

  v8f cst0 = {}, cst1 = {};        // cell state per column group, C-layout f32
  const v8f vzero = {};            // hoisted zero C-operand for WMMA

  #pragma unroll 1
  for (int t = 0; t < PRED; ++t) {
    const int tF = t * FSTRIDE;
    // ---- stage feature[:, HIST+t] (fp32 -> bf16) into cols 24..40 ----
    #pragma unroll
    for (int i = 0; i < 9; ++i) {
      if (foff[i] >= 0) {
        const float* fpt = feat + (size_t)(foff[i] + tF);
        xs[xoff[i]] = (bf16_t)(*fpt);
        if (t + 1 < PRED) __builtin_prefetch(fpt + FSTRIDE, 0, 1);
      }
    }

    // ---- single fused gate GEMM: [x(41)|h(32)] K=96, N=128, 24 WMMA ----
    v16bf a0 = lds_load_A(xs,  0, lane);
    v16bf a1 = lds_load_A(xs, 32, lane);
    v16bf a2 = lds_load_A(xs, 64, lane);

    v8f hbuf0, hbuf1;
    #pragma unroll
    for (int g2 = 0; g2 < 2; ++g2) {          // column group: tiles g2,2+g2,4+g2,6+g2
      v8f acc[4];
      #pragma unroll
      for (int q = 0; q < 4; ++q) {
        const int nt = 2 * q + g2;            // i/f/g/o tile for this group
        v16bf b0 = lds_load_B(&s_wg[0][0], nt * 16,  0, lane);
        v16bf b1 = lds_load_B(&s_wg[0][0], nt * 16, 32, lane);
        v16bf b2 = lds_load_B(&s_wg[0][0], nt * 16, 64, lane);
        acc[q] = WMMA_BF16(a0, b0, vzero);    // C = hoisted zero (bias added later)
        acc[q] = WMMA_BF16(a1, b1, acc[q]);
        acc[q] = WMMA_BF16(a2, b2, acc[q]);
      }
      // LSTM pointwise for this column group (bias folded in here)
      v8f& cst = (g2 == 0) ? cst0 : cst1;
      v8f& hb  = (g2 == 0) ? hbuf0 : hbuf1;
      #pragma unroll
      for (int v = 0; v < 8; ++v) {
        float ig = fast_sigmoid(acc[0][v] + bias_g[g2 + 0]);
        float fg = fast_sigmoid(acc[1][v] + bias_g[g2 + 2]);
        float gg = TANHF       (acc[2][v] + bias_g[g2 + 4]);
        float og = fast_sigmoid(acc[3][v] + bias_g[g2 + 6]);
        cst[v] = fg * cst[v] + ig * gg;
        hb[v]  = og * TANHF(cst[v]);
      }
      SCHED_FENCE();   // keep the two column groups' live ranges disjoint
    }

    // ---- recurrent h -> xs (K-interleaved): one pk-cvt + b32 store per pair ----
    #pragma unroll
    for (int v = 0; v < 8; ++v) {
      bf16x2 pr;
      pr.lo = (bf16_t)hbuf0[v];
      pr.hi = (bf16_t)hbuf1[v];
      *(bf16x2*)(xs + (rhalf + v) * XK + 48 + 2 * ncol) = pr;
    }

    // ---- fc_out: pred = h @ Wout^T + bo, reduce 16 lanes (cols) ----
    v8f p;
    #pragma unroll
    for (int v = 0; v < 8; ++v) p[v] = hbuf0[v] * wo0 + hbuf1[v] * wo1;
    #pragma unroll
    for (int m = 8; m >= 1; m >>= 1) {
      #pragma unroll
      for (int v = 0; v < 8; ++v) p[v] += __shfl_xor(p[v], m, 32);
    }

    // ---- shift xn window left (read phase, then write; DS in-order) ----
    bf16_t tmpv[12];
    #pragma unroll
    for (int i = 0; i < 12; ++i) if (sh_rd[i] >= 0) tmpv[i] = xs[sh_rd[i]];
    #pragma unroll
    for (int i = 0; i < 12; ++i) if (sh_rd[i] >= 0) xs[sh_rd[i] - 1] = tmpv[i];

    // ---- emit prediction + append to history col 23 ----
    const int tC = t * CCH;
    if (ncol == 0) {                           // lanes 0 (rows 0..7), 16 (rows 8..15)
      #pragma unroll
      for (int v = 0; v < 8; ++v) {
        float pv = p[v] + bo;
        out[obase[v] + tC] = pv;
        xs[(rhalf + v) * XK + 23] = (bf16_t)pv;
      }
    }
  }
}

extern "C" void kernel_launch(void* const* d_in, const int* in_sizes, int n_in,
                              void* d_out, int out_size, void* d_ws, size_t ws_size,
                              hipStream_t stream) {
  (void)in_sizes; (void)n_in; (void)out_size; (void)d_ws; (void)ws_size;
  const float* pm25 = (const float*)d_in[0];
  const float* feat = (const float*)d_in[1];
  // d_in[2] = time_feature (unused, matching the reference)
  const float* Win  = (const float*)d_in[3];
  const float* bin  = (const float*)d_in[4];
  const float* Wih  = (const float*)d_in[5];
  const float* Whh  = (const float*)d_in[6];
  const float* bih  = (const float*)d_in[7];
  const float* bhh  = (const float*)d_in[8];
  const float* Wout = (const float*)d_in[9];
  const float* bout = (const float*)d_in[10];
  float* out = (float*)d_out;

  dim3 grid(NROWS / 64);   // 1472 blocks x 128 threads (4 waves, 1 tile each)
  lstm_fused_kernel<<<grid, dim3(128), 0, stream>>>(
      pm25, feat, Win, bin, Wih, Whh, bih, bhh, Wout, bout, out);
}
